// MultiheadAttention_v6_21603685499632
// MI455X (gfx1250) — matrix operand order repaired and checked by reading, not compiled, not run on hardware
//
#include <hip/hip_runtime.h>
#include <hip/hip_bf16.h>

// ---------------------------------------------------------------------------
// CDNA5 wave32 WMMA types
// ---------------------------------------------------------------------------
typedef __attribute__((ext_vector_type(16))) __bf16 v16bf;
typedef __attribute__((ext_vector_type(8)))  float  v8f;

union FragBF {
    v16bf v;
    __bf16 h[16];
    uint4  q2[2];          // two 16B chunks -> direct b128 loads, no v_mov_b16
};

union Pack8 {
    uint4  u4;
    __bf16 h[8];
};

__device__ __forceinline__ v8f wmma_bf16(const FragBF& a, const FragBF& b, v8f c) {
    return __builtin_amdgcn_wmma_f32_16x16x32_bf16(false, a.v, false, b.v,
                                                   (short)0, c, false, false);
}

// Fragment-order layouts (ISA 7.12.2, wave32):
//  A (16-bit, 16x32):  lane g=lane>>4, row=lane&15 ; j=0..15 -> k = g*8 + j + (j>=8?8:0)
//  B (16-bit, 32x16):  lane g=lane>>4, n=lane&15   ; j=0..15 -> k = g*8 + j + (j>=8?8:0)
//  C/D (f32 16x16)  :  vgpr r, lane<16 -> (M=r,   N=lane)
//                              lane>=16 -> (M=r+8, N=lane-16)

// ---------------------------------------------------------------------------
// Kernel 0: one-shot prepack of Wq/Wk/Wv into bf16, B-fragment order.
// wpk[((mat*16 + ct)*8 + kc)*32 + lane][j] = W[k][n],
//   k = kc*32 + (j>>3)*16 + (lane>>4)*8 + (j&7), n = ct*16 + (lane&15)
// ---------------------------------------------------------------------------
__global__ void prepack_w(const float* __restrict__ Wq,
                          const float* __restrict__ Wk,
                          const float* __restrict__ Wv,
                          __bf16* __restrict__ wpk)
{
    const int E = 256;
    int base = (blockIdx.x * blockDim.x + threadIdx.x) * 4;
#pragma unroll
    for (int u = 0; u < 4; ++u) {
        int i = base + u;                       // < 3*65536
        int j    = i & 15;
        int lane = (i >> 4) & 31;
        int kc   = (i >> 9) & 7;
        int ct   = (i >> 12) & 15;
        int mat  = i >> 16;
        int k = kc * 32 + ((j >> 3) << 4) + ((lane >> 4) << 3) + (j & 7);
        int n = ct * 16 + (lane & 15);
        const float* W = (mat == 0) ? Wq : (mat == 1) ? Wk : Wv;
        wpk[i] = (__bf16)W[(size_t)k * E + n];
    }
}

// ---------------------------------------------------------------------------
// Kernel 1: q/k/v projections. grid=(P/16, 3), block=256 (8 waves).
// A tile (16 rows x 256) converted once to LDS in A-fragment order;
// B fragments are two b128 global loads from prepacked weights.
// ---------------------------------------------------------------------------
__global__ void proj_kernel(const float* __restrict__ query,
                            const float* __restrict__ key_x,
                            const float* __restrict__ value,
                            const float* __restrict__ b_q,
                            const float* __restrict__ b_k,
                            const float* __restrict__ b_v,
                            const __bf16* __restrict__ wpk,
                            float* __restrict__ qbuf,
                            float* __restrict__ kbuf,
                            float* __restrict__ vbuf)
{
    const int E = 256;
    const int mat = blockIdx.y;
    const float* X    = (mat == 0) ? query : (mat == 1) ? key_x : value;
    const float* bias = (mat == 0) ? b_q   : (mat == 1) ? b_k   : b_v;
    float*       out  = (mat == 0) ? qbuf  : (mat == 1) ? kbuf  : vbuf;

    __shared__ __bf16 lds_a[16 * 256];   // A-fragment order: [(kc*32+lane)*16 + j]

    const int t    = threadIdx.x;
    const int wave = t >> 5;
    const int lane = t & 31;
    const int m0   = blockIdx.x * 16;

    // ---- stage A tile (16x256) to LDS in fragment order -----------------
    {
        const int row   = t >> 4;
        const int kbase = (t & 15) * 16;
        const float4* X4 = (const float4*)(X + (size_t)(m0 + row) * E + kbase);
#pragma unroll
        for (int ss = 0; ss < 4; ++ss) {
            float4 f = X4[ss];
            float vals[4] = {f.x, f.y, f.z, f.w};
#pragma unroll
            for (int u = 0; u < 4; ++u) {
                int k  = kbase + ss * 4 + u;
                int kc = k >> 5;
                int km = k & 31;
                int g  = (km >> 3) & 1;
                int ln = g * 16 + row;
                int j  = ((km >> 4) << 3) | (km & 7);
                lds_a[(kc * 32 + ln) * 16 + j] = (__bf16)vals[u];
            }
        }
    }
    __syncthreads();

    // ---- WMMA main loop --------------------------------------------------
    v8f acc[2] = {};
    for (int kc = 0; kc < 8; ++kc) {
        FragBF a;
        const uint4* ap = (const uint4*)&lds_a[(kc * 32 + lane) * 16];
        a.q2[0] = ap[0];
        a.q2[1] = ap[1];
#pragma unroll
        for (int ci = 0; ci < 2; ++ci) {
            const int ct = wave + ci * 8;
            FragBF b;
            const uint4* bp = (const uint4*)&wpk[(((size_t)(mat * 16 + ct) * 8 + kc) * 32 + lane) * 16];
            b.q2[0] = bp[0];
            b.q2[1] = bp[1];
            acc[ci] = wmma_bf16(a, b, acc[ci]);
        }
    }

    const int nlan  = lane & 15;
    const int mbase = (lane < 16) ? 0 : 8;
#pragma unroll
    for (int ci = 0; ci < 2; ++ci) {
        const int n  = (wave + ci * 8) * 16 + nlan;
        const float bb = bias[n];
#pragma unroll
        for (int r = 0; r < 8; ++r)
            out[(size_t)(m0 + mbase + r) * E + n] = acc[ci][r] + bb;
    }
}

// ---------------------------------------------------------------------------
// Kernel 2: fused edge pipeline, one block per 4 points (= 128 edges).
// ---------------------------------------------------------------------------
__global__ void fused_edge_attn(const float* __restrict__ qbuf,
                                const float* __restrict__ kbuf,
                                const float* __restrict__ vbuf,
                                const int*   __restrict__ index_pair,
                                const float* __restrict__ raw,   // [P,N,H]
                                const float* __restrict__ rpe,   // [P,N,3]
                                const float* __restrict__ Wg1, const float* __restrict__ bg1,
                                const float* __restrict__ Wg2, const float* __restrict__ bg2,
                                float* __restrict__ out)
{
    const int E = 256;
    const int TP = 4;

    extern __shared__ char smem[];
    float*  q_s    = (float*)smem;                    // 1024 f32   (4096 B)
    int*    idx_s  = (int*)(q_s + 1024);              // 128        (512 B)
    float*  sh_s   = (float*)(idx_s + 128);           // 384 f32    (1536 B)
    float*  dot_s  = sh_s + 384;                      // 1024 f32   (4096 B)
    float*  att_s  = dot_s + 1024;                    // 1024 f32   (4096 B)
    float*  g2_s   = att_s + 1024;                    // 4096 f32   (16384 B)
    __bf16* diff_s = (__bf16*)(g2_s + 4096);          // 32768 bf16 (65536 B)
    __bf16* wg1_s  = diff_s + 32768;                  // 8192 bf16  (16384 B) fragment order
    __bf16* wg2_s  = wg1_s + 8192;                    // 1024 bf16  (2048 B)  fragment order
    __bf16* g1_s   = wg2_s + 1024;                    // 4096 bf16  (8192 B)  row-major [128][32]

    const int t    = threadIdx.x;
    const int lane = t & 31;
    const int wave = t >> 5;
    const int p0   = blockIdx.x * TP;

    // ---- staging --------------------------------------------------------
    ((float4*)q_s)[t] = ((const float4*)(qbuf + (size_t)p0 * E))[t];

    if (t < 128) {
        const int e = t;
        idx_s[e] = index_pair[(size_t)p0 * 32 + e];
        const size_t rb = ((size_t)p0 * 32 + e) * 3;
        float x = rpe[rb + 0], y = rpe[rb + 1], z = rpe[rb + 2];
        float len = sqrtf(x * x + y * y + z * z);
        float inv = 1.0f / fmaxf(len, 1e-12f);
        float tt  = (len - 0.001f) * (1.0f / 0.004f);
        tt = fminf(fmaxf(tt, 0.0f), 1.0f);
        float cut = 0.5f * (1.0f - __cosf(3.14159265358979323f * tt));
        float s   = 1.7320508075688772f * inv * cut;
        sh_s[e * 3 + 0] = x * s;
        sh_s[e * 3 + 1] = y * s;
        sh_s[e * 3 + 2] = z * s;
    }
    // Wg1 in B-fragment order: [((nt*8 + kc)*32 + lane)*16 + j]
    for (int i = t; i < 8192; i += 256) {
        int j  = i & 15;
        int ln = (i >> 4) & 31;
        int kc = (i >> 9) & 7;
        int nt = i >> 12;
        int k = kc * 32 + ((j >> 3) << 4) + ((ln >> 4) << 3) + (j & 7);
        int n = nt * 16 + (ln & 15);
        wg1_s[i] = (n < 24) ? (__bf16)Wg1[(size_t)k * 24 + n] : (__bf16)0.0f;
    }
    // Wg2 in B-fragment order: [(nt*32 + lane)*16 + j]
    for (int i = t; i < 1024; i += 256) {
        int j  = i & 15;
        int ln = (i >> 4) & 31;
        int nt = i >> 9;
        int k = ((j >> 3) << 4) + ((ln >> 4) << 3) + (j & 7);
        int n = nt * 16 + (ln & 15);
        wg2_s[i] = (k < 24 && n < 24) ? (__bf16)Wg2[(size_t)k * 24 + n] : (__bf16)0.0f;
    }
    __syncthreads();

    // ---- gather k, diff(bf16, b128 stores), per-head dot logits ---------
    {
        const int e    = t >> 1;
        const int half = t & 1;
        const int li   = e >> 5;
        const int row  = idx_s[e];
        const float4* krow4 = (const float4*)(kbuf + (size_t)row * E + half * 128);
        const float4* qrow4 = (const float4*)(q_s + li * 256 + half * 128);
        uint4* drow4 = (uint4*)(diff_s + e * 256 + half * 128);
        __builtin_prefetch(vbuf + (size_t)row * E, 0, 1);   // global_prefetch_b8
        float dacc[4] = {0.f, 0.f, 0.f, 0.f};
#pragma unroll
        for (int c8 = 0; c8 < 16; ++c8) {
            float4 k0 = krow4[c8 * 2 + 0];
            float4 k1 = krow4[c8 * 2 + 1];
            float4 q0 = qrow4[c8 * 2 + 0];
            float4 q1 = qrow4[c8 * 2 + 1];
            Pack8 p;
            p.h[0] = (__bf16)(k0.x - q0.x); p.h[1] = (__bf16)(k0.y - q0.y);
            p.h[2] = (__bf16)(k0.z - q0.z); p.h[3] = (__bf16)(k0.w - q0.w);
            p.h[4] = (__bf16)(k1.x - q1.x); p.h[5] = (__bf16)(k1.y - q1.y);
            p.h[6] = (__bf16)(k1.z - q1.z); p.h[7] = (__bf16)(k1.w - q1.w);
            drow4[c8] = p.u4;
            dacc[c8 >> 2] += k0.x * q0.x + k0.y * q0.y + k0.z * q0.z + k0.w * q0.w
                           + k1.x * q1.x + k1.y * q1.y + k1.z * q1.z + k1.w * q1.w;
        }
#pragma unroll
        for (int hh = 0; hh < 4; ++hh)
            dot_s[e * 8 + half * 4 + hh] = dacc[hh];
    }
    __syncthreads();

    const int nlan  = lane & 15;
    const int kbA   = (lane >> 4) * 8;
    const int mbase = (lane < 16) ? 0 : 8;
    const int m0    = wave * 16;
    const int mrow  = m0 + (lane & 15);

    // ---- GEMM1: diff[128x256] @ Wg1 -> relu -> g1 (bf16) ----------------
    {
        v8f acc[2] = {};
        for (int kc = 0; kc < 8; ++kc) {
            FragBF a;
            const uint4* ap = (const uint4*)&diff_s[mrow * 256 + kc * 32 + kbA];
            a.q2[0] = ap[0];            // k = +0..7
            a.q2[1] = ap[2];            // k = +16..23
#pragma unroll
            for (int nt = 0; nt < 2; ++nt) {
                FragBF b;
                const uint4* bp = (const uint4*)&wg1_s[((nt * 8 + kc) * 32 + lane) * 16];
                b.q2[0] = bp[0];
                b.q2[1] = bp[1];
                acc[nt] = wmma_bf16(a, b, acc[nt]);
            }
        }
#pragma unroll
        for (int nt = 0; nt < 2; ++nt) {
            const int n = nt * 16 + nlan;
            const float bb = (n < 24) ? bg1[n] : 0.0f;
#pragma unroll
            for (int r = 0; r < 8; ++r) {
                float v = fmaxf(acc[nt][r] + bb, 0.0f);
                g1_s[(m0 + mbase + r) * 32 + n] = (__bf16)v;
            }
        }
    }
    __syncthreads();

    // ---- GEMM2: g1[128x32] @ Wg2 (+bg2) -> g2 (f32) ---------------------
    {
        v8f acc[2] = {};
        FragBF a;
        const uint4* ap0 = (const uint4*)&g1_s[mrow * 32 + kbA];
        const uint4* ap1 = (const uint4*)&g1_s[mrow * 32 + kbA + 16];
        a.q2[0] = ap0[0];
        a.q2[1] = ap1[0];
#pragma unroll
        for (int nt = 0; nt < 2; ++nt) {
            FragBF b;
            const uint4* bp = (const uint4*)&wg2_s[(nt * 32 + lane) * 16];
            b.q2[0] = bp[0];
            b.q2[1] = bp[1];
            acc[nt] = wmma_bf16(a, b, acc[nt]);
        }
#pragma unroll
        for (int nt = 0; nt < 2; ++nt) {
            const int n = nt * 16 + nlan;
            const float bb = (n < 24) ? bg2[n] : 0.0f;
#pragma unroll
            for (int r = 0; r < 8; ++r)
                g2_s[(m0 + mbase + r) * 32 + n] = acc[nt][r] + bb;
        }
    }
    __syncthreads();

    // ---- logits + softmax over the 32 neighbors -------------------------
    if (t < 32) {
        const int li = t >> 3, h = t & 7;
        const int p  = p0 + li;
        float lg[32];
        float mx = -1e30f;
#pragma unroll
        for (int n = 0; n < 32; ++n) {
            const int e = li * 32 + n;
            float pos = g2_s[e * 32 + h * 3 + 0] * sh_s[e * 3 + 0]
                      + g2_s[e * 32 + h * 3 + 1] * sh_s[e * 3 + 1]
                      + g2_s[e * 32 + h * 3 + 2] * sh_s[e * 3 + 2];
            float l = (dot_s[e * 8 + h] + pos + raw[((size_t)p * 32 + n) * 8 + h])
                      * 0.17677669529663687f;   // 1/sqrt(32)
            lg[n] = l;
            mx = fmaxf(mx, l);
        }
        float sum = 0.0f;
#pragma unroll
        for (int n = 0; n < 32; ++n) { float ex = __expf(lg[n] - mx); lg[n] = ex; sum += ex; }
        const float inv = 1.0f / sum;
#pragma unroll
        for (int n = 0; n < 32; ++n) att_s[(li * 32 + n) * 8 + h] = lg[n] * inv;
    }
    __syncthreads();

    // ---- out[p] = sum_n atten * v_gathered (float4-wide) ----------------
    {
        const int li = t >> 6;
        const int c4 = (t & 63) * 4;
        const int h  = (t & 63) >> 3;
        float4 acc = {0.f, 0.f, 0.f, 0.f};
#pragma unroll 4
        for (int n = 0; n < 32; ++n) {
            const int e = li * 32 + n;
            const float a = att_s[e * 8 + h];
            float4 v = *(const float4*)(vbuf + (size_t)idx_s[e] * E + c4);
            acc.x += a * v.x; acc.y += a * v.y; acc.z += a * v.z; acc.w += a * v.w;
        }
        *(float4*)(out + (size_t)(p0 + li) * E + c4) = acc;
    }
}

// ---------------------------------------------------------------------------
// Host launch
// ---------------------------------------------------------------------------
extern "C" void kernel_launch(void* const* d_in, const int* in_sizes, int n_in,
                              void* d_out, int out_size, void* d_ws, size_t ws_size,
                              hipStream_t stream)
{
    (void)in_sizes; (void)n_in; (void)out_size; (void)ws_size;
    const int P = 8192, E = 256;

    const float* query = (const float*)d_in[0];
    const float* key_x = (const float*)d_in[1];
    const float* value = (const float*)d_in[2];
    const int*   idxp  = (const int*)  d_in[3];
    const float* raw   = (const float*)d_in[4];
    const float* rpe   = (const float*)d_in[5];
    const float* Wq    = (const float*)d_in[6];
    const float* bq    = (const float*)d_in[7];
    const float* Wk    = (const float*)d_in[8];
    const float* bk    = (const float*)d_in[9];
    const float* Wv    = (const float*)d_in[10];
    const float* bv    = (const float*)d_in[11];
    const float* Wg1   = (const float*)d_in[12];
    const float* bg1   = (const float*)d_in[13];
    const float* Wg2   = (const float*)d_in[14];
    const float* bg2   = (const float*)d_in[15];
    float*       out   = (float*)d_out;

    float*  qbuf = (float*)d_ws;
    float*  kbuf = qbuf + (size_t)P * E;
    float*  vbuf = kbuf + (size_t)P * E;
    __bf16* wpk  = (__bf16*)(vbuf + (size_t)P * E);   // 3*65536 bf16

    // 0) prepack weights (3*65536 elements, 4 per thread)
    prepack_w<<<192, 256, 0, stream>>>(Wq, Wk, Wv, wpk);

    // 1) projections
    dim3 gp(P / 16, 3);
    proj_kernel<<<gp, 256, 0, stream>>>(query, key_x, value,
                                        bq, bk, bv, wpk,
                                        qbuf, kbuf, vbuf);

    // 2) fused edge attention
    // LDS: q 4096 + idx 512 + sh 1536 + dot 4096 + att 4096 + g2 16384
    //      + diff 65536 + wg1 16384 + wg2 2048 + g1 8192 = 122880 bytes
    const size_t smem_bytes = 122880;
    fused_edge_attn<<<P / 4, 256, smem_bytes, stream>>>(qbuf, kbuf, vbuf,
                                                        idxp, raw, rpe,
                                                        Wg1, bg1, Wg2, bg2,
                                                        out);
}
